// GroupedQueryAttention_21844203668225
// MI455X (gfx1250) — matrix operand order repaired and checked by reading, not compiled, not run on hardware
//
#include <hip/hip_runtime.h>

#define D_MODEL   2048
#define T_SEQ     2048
#define BATCH     2
#define NH        16
#define NKV       4
#define HD        128
#define BT        (BATCH * T_SEQ)     // 4096 tokens
#define EPS_RMS   1e-5f

typedef __attribute__((ext_vector_type(16))) __bf16 v16bf;
typedef __attribute__((ext_vector_type(8)))  float  v8f;
typedef __attribute__((ext_vector_type(4)))  float  v4f;

// ---------- helpers ----------
__device__ __forceinline__ unsigned short f2bf_u16(float f) {
  unsigned u = __float_as_uint(f);
  unsigned r = u + 0x7FFFu + ((u >> 16) & 1u);   // round-to-nearest-even
  return (unsigned short)(r >> 16);
}
__device__ __forceinline__ float bfu_to_f(unsigned short h) {
  return __uint_as_float(((unsigned)h) << 16);
}
__device__ __forceinline__ __bf16 bfbits(unsigned short h) {
  return __builtin_bit_cast(__bf16, h);
}
// split f32 into bf16 hi + bf16 lo (residual), for bf16x3 fp32-accurate GEMM
__device__ __forceinline__ void split_bf16(float x, __bf16 &hi, __bf16 &lo) {
  unsigned short h = f2bf_u16(x);
  hi = bfbits(h);
  lo = bfbits(f2bf_u16(x - bfu_to_f(h)));
}
__device__ __forceinline__ v8f zero8() {
  v8f z;
#pragma unroll
  for (int i = 0; i < 8; ++i) z[i] = 0.0f;
  return z;
}

// =====================================================================
// GEMM: C[M,N] = A[M,K] @ B[K,N], fp32 in/out, bf16x3 WMMA internally.
// Workgroup: 256 threads (8 waves). Tile: 128(M) x 64(N), K-chunk 32.
// LDS holds A/B tiles pre-swizzled into WMMA fragment layout.
// =====================================================================
#define GM 128
#define GN 64
#define GK 32

__global__ __launch_bounds__(256, 1)
void gemm_bf16x3(const float* __restrict__ A, const float* __restrict__ B,
                 float* __restrict__ C, int M, int N, int K) {
  __shared__ __align__(32) __bf16 aHi[GM * GK];   // (strip*32+lane)*16+j
  __shared__ __align__(32) __bf16 aLo[GM * GK];
  __shared__ __align__(32) __bf16 bHi[GK * GN];   // (ntile*32+lane)*16+j
  __shared__ __align__(32) __bf16 bLo[GK * GN];

  const int tid  = threadIdx.x;
  const int wave = tid >> 5;
  const int lane = tid & 31;
  const int tileM = blockIdx.y * GM;
  const int tileN = blockIdx.x * GN;

  v8f acc[4];
#pragma unroll
  for (int n = 0; n < 4; ++n) acc[n] = zero8();

  for (int k0 = 0; k0 < K; k0 += GK) {
    __syncthreads();
    // ---- A tile 128x32 (1024 float4, 4/thread), split & swizzle to A-frag layout
#pragma unroll
    for (int i = 0; i < 4; ++i) {
      int fidx = tid + i * 256;
      int row  = fidx >> 3;
      int c4   = (fidx & 7) << 2;
      v4f v = *(const v4f*)(A + (size_t)(tileM + row) * K + k0 + c4);
      int strip = row >> 4, Mloc = row & 15;
#pragma unroll
      for (int cc = 0; cc < 4; ++cc) {
        int kk = c4 + cc;
        int l  = Mloc + (((kk >> 3) & 1) << 4);
        int j  = kk - ((kk >= 16) ? 8 : 0) - (((kk >> 3) & 1) << 3);
        int idx = ((strip << 5) + l) * 16 + j;
        __bf16 hh, ll; split_bf16(v[cc], hh, ll);
        aHi[idx] = hh; aLo[idx] = ll;
      }
    }
    // ---- B tile 32x64 (512 float4, 2/thread) to B-frag layout
#pragma unroll
    for (int i = 0; i < 2; ++i) {
      int fidx = tid + i * 256;
      int row  = fidx >> 4;          // K index 0..31
      int c4   = (fidx & 15) << 2;   // N offset
      v4f v = *(const v4f*)(B + (size_t)(k0 + row) * N + tileN + c4);
#pragma unroll
      for (int cc = 0; cc < 4; ++cc) {
        int nn = c4 + cc;
        int nt = nn >> 4, nl = nn & 15;
        int l  = (((row >> 3) & 1) << 4) + nl;
        int j  = (row & 7) + ((row >> 4) << 3);
        int idx = ((nt << 5) + l) * 16 + j;
        __bf16 hh, ll; split_bf16(v[cc], hh, ll);
        bHi[idx] = hh; bLo[idx] = ll;
      }
    }
    __syncthreads();
    // ---- compute: wave w owns M-strip w; 4 N-tiles; 3 WMMAs each (hi*hi + hi*lo + lo*hi)
    v16bf aH = *(const v16bf*)&aHi[((wave << 5) + lane) * 16];
    v16bf aL = *(const v16bf*)&aLo[((wave << 5) + lane) * 16];
#pragma unroll
    for (int n = 0; n < 4; ++n) {
      v16bf bH = *(const v16bf*)&bHi[((n << 5) + lane) * 16];
      v16bf bL = *(const v16bf*)&bLo[((n << 5) + lane) * 16];
      acc[n] = __builtin_amdgcn_wmma_f32_16x16x32_bf16(false, aH, false, bH, (short)0, acc[n], false, false);
      acc[n] = __builtin_amdgcn_wmma_f32_16x16x32_bf16(false, aH, false, bL, (short)0, acc[n], false, false);
      acc[n] = __builtin_amdgcn_wmma_f32_16x16x32_bf16(false, aL, false, bH, (short)0, acc[n], false, false);
    }
  }
  // ---- store C (C-frag layout: lane -> N=lane%16, M=(lane/16)*8+i)
  const int lh = lane >> 4, nl = lane & 15;
#pragma unroll
  for (int n = 0; n < 4; ++n)
#pragma unroll
    for (int i = 0; i < 8; ++i) {
      int row = tileM + (wave << 4) + (lh << 3) + i;
      int col = tileN + (n << 4) + nl;
      C[(size_t)row * N + col] = acc[n][i];
    }
}

// =====================================================================
// Per-head RMSNorm + RoPE. One wave32 per 128-wide row.
// in:  [B*T, Hn*128]  (row = b*T+t, col = h*128+d)
// out: [B, Hn, T, 128]
// =====================================================================
__global__ __launch_bounds__(256, 1)
void normrope(const float* __restrict__ in, const float* __restrict__ w,
              float* __restrict__ out, int Hn) {
  const int row  = blockIdx.x * 8 + (threadIdx.x >> 5);
  const int lane = threadIdx.x & 31;
  const int t = row % T_SEQ;
  const int h = (row / T_SEQ) % Hn;
  const int b = row / (T_SEQ * Hn);
  const float* src = in + (size_t)(b * T_SEQ + t) * (Hn * HD) + h * HD;

  float x[4]; float ss = 0.f;
#pragma unroll
  for (int ii = 0; ii < 4; ++ii) { x[ii] = src[lane + ii * 32]; ss += x[ii] * x[ii]; }
#pragma unroll
  for (int m = 16; m >= 1; m >>= 1) ss += __shfl_xor(ss, m, 32);
  const float rms = rsqrtf(ss * (1.0f / HD) + EPS_RMS);

  // inv_freq[j] = 500000^(-j/64) = exp(c0*j), c0 = -ln(500000)/64
  const float c0 = -0.20503692777638263f;
  float th0 = (float)t * expf(c0 * (float)lane);
  float th1 = (float)t * expf(c0 * (float)(lane + 32));
  float sn0, cs0, sn1, cs1;
  sincosf(th0, &sn0, &cs0);
  sincosf(th1, &sn1, &cs1);

  float xn[4];
#pragma unroll
  for (int ii = 0; ii < 4; ++ii) xn[ii] = x[ii] * rms * w[lane + ii * 32];

  float* dst = out + (size_t)row * HD;
  dst[lane]      = xn[0] * cs0 - xn[2] * sn0;
  dst[lane + 32] = xn[1] * cs1 - xn[3] * sn1;
  dst[lane + 64] = xn[2] * cs0 + xn[0] * sn0;
  dst[lane + 96] = xn[3] * cs1 + xn[1] * sn1;
}

// =====================================================================
// Causal flash attention with GQA (4 Q heads share 1 KV head).
// Workgroup: 256 threads = 8 waves; Q tile 128 rows (16 rows/wave);
// KV blocks of 64. bf16x3 WMMA for both Q@K^T and P@V.
// LDS (64KB): K-frags hi/lo (32KB, reused as P stage) + V-frags hi/lo (32KB)
// =====================================================================
#define FBM 128
#define FBN 64

__global__ __launch_bounds__(256, 1)
void flash_gqa(const float* __restrict__ Qn,   // [B,NH,T,HD]
               const float* __restrict__ Kn,   // [B,NKV,T,HD]
               const float* __restrict__ Vp,   // [B*T, NKV*HD]
               float* __restrict__ Out) {      // [B*T, NH*HD]
  __shared__ __align__(32) char smem[64 * 1024];
  __bf16* kHi = (__bf16*)(smem);               // 8192 el (4c x 4n frags)
  __bf16* kLo = (__bf16*)(smem + 16 * 1024);
  __bf16* vHi = (__bf16*)(smem + 32 * 1024);   // 8192 el (2c x 8n frags)
  __bf16* vLo = (__bf16*)(smem + 48 * 1024);
  float*  stage = (float*)(smem);              // 8 waves x 16x64 f32 = 32KB (over K region)

  const int tid = threadIdx.x, wave = tid >> 5, lane = tid & 31;
  const int qt = blockIdx.x, h = blockIdx.y, b = blockIdx.z;
  const int hv = h >> 2;
  const int qbase = qt * FBM;
  const int lh = lane >> 4, nl = lane & 15;

  // ---- load Q strip (16 rows x 128) into A-frags, pre-scaled by 1/sqrt(HD)
  const float scale = 0.08838834764831845f;
  const int qrow = qbase + (wave << 4) + nl;    // M = lane%16
  const float* qsrc = Qn + ((size_t)(b * NH + h) * T_SEQ + qrow) * HD;
  v16bf qH[4], qL[4];
#pragma unroll
  for (int c = 0; c < 4; ++c) {
    int d0 = c * 32 + (lh << 3);
    v4f t[4];
    t[0] = *(const v4f*)(qsrc + d0);
    t[1] = *(const v4f*)(qsrc + d0 + 4);
    t[2] = *(const v4f*)(qsrc + d0 + 16);
    t[3] = *(const v4f*)(qsrc + d0 + 20);
#pragma unroll
    for (int j = 0; j < 16; ++j) {
      __bf16 hh, ll; split_bf16(t[j >> 2][j & 3] * scale, hh, ll);
      qH[c][j] = hh; qL[c][j] = ll;
    }
  }

  v8f o[8];
#pragma unroll
  for (int n = 0; n < 8; ++n) o[n] = zero8();
  float mrow[8], lrow[8];
#pragma unroll
  for (int i = 0; i < 8; ++i) { mrow[i] = -1e30f; lrow[i] = 0.f; }

  const int nblocks = (qbase + FBM) / FBN;
  for (int kb = 0; kb < nblocks; ++kb) {
    const int kv0 = kb * FBN;
    __syncthreads();   // prev iter's stage/V reads done before overwrite
    // ---- cooperative K block (64x128) -> B-frag layout (contraction = d)
#pragma unroll
    for (int i = 0; i < 8; ++i) {
      int fidx = tid + i * 256;
      int r  = fidx >> 5;            // kv row in block (N dim of K^T)
      int c4 = (fidx & 31) << 2;     // d offset
      v4f v = *(const v4f*)(Kn + ((size_t)(b * NKV + hv) * T_SEQ + kv0 + r) * HD + c4);
      int nt = r >> 4, nnl = r & 15;
#pragma unroll
      for (int cc = 0; cc < 4; ++cc) {
        int dk = c4 + cc;
        int ch = dk >> 5, kk = dk & 31;
        int l  = (((kk >> 3) & 1) << 4) + nnl;
        int j  = (kk & 7) + ((kk >> 4) << 3);
        int idx = (((ch << 2) + nt) * 32 + l) * 16 + j;
        __bf16 hh, ll; split_bf16(v[cc], hh, ll);
        kHi[idx] = hh; kLo[idx] = ll;
      }
    }
    // ---- cooperative V block (64x128) -> B-frag layout (contraction = kv)
#pragma unroll
    for (int i = 0; i < 8; ++i) {
      int fidx = tid + i * 256;
      int r  = fidx >> 5;            // kv row = contraction index
      int c4 = (fidx & 31) << 2;     // d offset = N dim
      v4f v = *(const v4f*)(Vp + ((size_t)(b * T_SEQ) + kv0 + r) * (NKV * HD) + hv * HD + c4);
      int ch = r >> 5, kk = r & 31;
#pragma unroll
      for (int cc = 0; cc < 4; ++cc) {
        int d  = c4 + cc;
        int nt = d >> 4, dnl = d & 15;
        int l  = (((kk >> 3) & 1) << 4) + dnl;
        int j  = (kk & 7) + ((kk >> 4) << 3);
        int idx = (((ch << 3) + nt) * 32 + l) * 16 + j;
        __bf16 hh, ll; split_bf16(v[cc], hh, ll);
        vHi[idx] = hh; vLo[idx] = ll;
      }
    }
    __syncthreads();

    // ---- S = Q @ K^T (bf16x3)
    v8f s[4];
#pragma unroll
    for (int n = 0; n < 4; ++n) s[n] = zero8();
#pragma unroll
    for (int c = 0; c < 4; ++c) {
#pragma unroll
      for (int n = 0; n < 4; ++n) {
        v16bf bH = *(const v16bf*)&kHi[(((c << 2) + n) * 32 + lane) * 16];
        v16bf bL = *(const v16bf*)&kLo[(((c << 2) + n) * 32 + lane) * 16];
        s[n] = __builtin_amdgcn_wmma_f32_16x16x32_bf16(false, qH[c], false, bH, (short)0, s[n], false, false);
        s[n] = __builtin_amdgcn_wmma_f32_16x16x32_bf16(false, qH[c], false, bL, (short)0, s[n], false, false);
        s[n] = __builtin_amdgcn_wmma_f32_16x16x32_bf16(false, qL[c], false, bH, (short)0, s[n], false, false);
      }
    }
    // ---- causal mask
    const int qr0 = qbase + (wave << 4) + (lh << 3);
#pragma unroll
    for (int n = 0; n < 4; ++n) {
      int kvidx = kv0 + (n << 4) + nl;
#pragma unroll
      for (int i = 0; i < 8; ++i)
        if (kvidx > qr0 + i) s[n][i] = -1e30f;
    }
    // ---- online softmax (rows live in 16-lane halves; xor<=8 reductions)
    float alpha[8];
#pragma unroll
    for (int i = 0; i < 8; ++i) {
      float v = fmaxf(fmaxf(s[0][i], s[1][i]), fmaxf(s[2][i], s[3][i]));
#pragma unroll
      for (int m = 8; m >= 1; m >>= 1) v = fmaxf(v, __shfl_xor(v, m, 32));
      float mnew = fmaxf(mrow[i], v);
      alpha[i] = __expf(mrow[i] - mnew);
      mrow[i] = mnew;
    }
    float rsum[8];
#pragma unroll
    for (int i = 0; i < 8; ++i) rsum[i] = 0.f;
#pragma unroll
    for (int n = 0; n < 4; ++n)
#pragma unroll
      for (int i = 0; i < 8; ++i) {
        float p = __expf(s[n][i] - mrow[i]);
        s[n][i] = p;
        rsum[i] += p;
      }
#pragma unroll
    for (int i = 0; i < 8; ++i) {
      float v = rsum[i];
#pragma unroll
      for (int m = 8; m >= 1; m >>= 1) v += __shfl_xor(v, m, 32);
      lrow[i] = lrow[i] * alpha[i] + v;
    }
#pragma unroll
    for (int n = 0; n < 8; ++n)
#pragma unroll
      for (int i = 0; i < 8; ++i) o[n][i] *= alpha[i];

    __syncthreads();   // everyone done reading K-frags; reuse as P stage
    // ---- stage P (C-layout) as f32, re-read in A-layout (per-wave slice)
    float* st = stage + wave * 1024;
#pragma unroll
    for (int n = 0; n < 4; ++n)
#pragma unroll
      for (int i = 0; i < 8; ++i)
        st[((lh << 3) + i) * 64 + (n << 4) + nl] = s[n][i];
    // same-wave LDS ops are in-order; loads below see the stores above
#pragma unroll
    for (int c = 0; c < 2; ++c) {
      v16bf pH, pL;
#pragma unroll
      for (int j = 0; j < 16; ++j) {
        int kk = c * 32 + (lh << 3) + (j < 8 ? j : j + 8);
        float pv = st[nl * 64 + kk];
        __bf16 hh, ll; split_bf16(pv, hh, ll);
        pH[j] = hh; pL[j] = ll;
      }
#pragma unroll
      for (int n = 0; n < 8; ++n) {
        v16bf bH = *(const v16bf*)&vHi[(((c << 3) + n) * 32 + lane) * 16];
        v16bf bL = *(const v16bf*)&vLo[(((c << 3) + n) * 32 + lane) * 16];
        o[n] = __builtin_amdgcn_wmma_f32_16x16x32_bf16(false, pH, false, bH, (short)0, o[n], false, false);
        o[n] = __builtin_amdgcn_wmma_f32_16x16x32_bf16(false, pH, false, bL, (short)0, o[n], false, false);
        o[n] = __builtin_amdgcn_wmma_f32_16x16x32_bf16(false, pL, false, bH, (short)0, o[n], false, false);
      }
    }
  }
  // ---- finalize: O /= l, scatter to [B*T, NH*HD]
#pragma unroll
  for (int i = 0; i < 8; ++i) lrow[i] = 1.0f / lrow[i];
  const int orow = qbase + (wave << 4) + (lh << 3);
#pragma unroll
  for (int n = 0; n < 8; ++n)
#pragma unroll
    for (int i = 0; i < 8; ++i) {
      int row = b * T_SEQ + orow + i;
      int col = h * HD + (n << 4) + nl;
      Out[(size_t)row * (NH * HD) + col] = o[n][i] * lrow[i];
    }
}

// =====================================================================
// Host launcher
// =====================================================================
extern "C" void kernel_launch(void* const* d_in, const int* in_sizes, int n_in,
                              void* d_out, int out_size, void* d_ws, size_t ws_size,
                              hipStream_t stream) {
  (void)in_sizes; (void)n_in; (void)out_size; (void)ws_size;
  const float* x  = (const float*)d_in[0];
  const float* Wq = (const float*)d_in[1];
  const float* Wk = (const float*)d_in[2];
  const float* Wv = (const float*)d_in[3];
  const float* Wo = (const float*)d_in[4];
  const float* qw = (const float*)d_in[5];
  const float* kw = (const float*)d_in[6];
  float* out = (float*)d_out;

  float* ws = (float*)d_ws;
  float* Qp = ws;                                   //  8M f32: [BT,2048]
  float* Kp = Qp + (size_t)BT * 2048;               //  2M f32: [BT,512]
  float* Vp = Kp + (size_t)BT * 512;                //  2M f32: [BT,512]
  float* Qn = Vp + (size_t)BT * 512;                //  8M f32: [B,NH,T,HD]
  float* Kv = Qn + (size_t)BT * 2048;               //  2M f32: [B,NKV,T,HD]
  float* AO = Qp;                                   // alias: Qp dead after normrope

  dim3 blk(256);
  // QKV projections (bf16x3 WMMA GEMM)
  gemm_bf16x3<<<dim3(2048 / GN, BT / GM), blk, 0, stream>>>(x, Wq, Qp, BT, 2048, D_MODEL);
  gemm_bf16x3<<<dim3(512 / GN, BT / GM),  blk, 0, stream>>>(x, Wk, Kp, BT, 512, D_MODEL);
  gemm_bf16x3<<<dim3(512 / GN, BT / GM),  blk, 0, stream>>>(x, Wv, Vp, BT, 512, D_MODEL);
  // per-head RMSNorm + RoPE
  normrope<<<dim3((BATCH * NH * T_SEQ) / 8),  blk, 0, stream>>>(Qp, qw, Qn, NH);
  normrope<<<dim3((BATCH * NKV * T_SEQ) / 8), blk, 0, stream>>>(Kp, kw, Kv, NKV);
  // causal flash attention with GQA
  flash_gqa<<<dim3(T_SEQ / FBM, NH, BATCH), blk, 0, stream>>>(Qn, Kv, Vp, AO);
  // output projection
  gemm_bf16x3<<<dim3(2048 / GN, BT / GM), blk, 0, stream>>>(AO, Wo, out, BT, 2048, 2048);
}
